// ConvParaLIF_41059887349806
// MI455X (gfx1250) — matrix operand order repaired and checked by reading, not compiled, not run on hardware
//
#include <hip/hip_runtime.h>
#include <hip/hip_fp16.h>
#include <math.h>

typedef __attribute__((ext_vector_type(16))) _Float16 v16h;
typedef __attribute__((ext_vector_type(8)))  _Float16 v8h;
typedef __attribute__((ext_vector_type(4)))  _Float16 v4h;
typedef __attribute__((ext_vector_type(8)))  float    v8f;

#define BSZ 16
#define CCH 128
#define HSP 112
#define WSP 112
#define ALPHA_C 0.36787944117144233f
#define BETA_C  0.36787944117144233f

#define XP 136                    // ci pitch inside x tile (bank pad, keeps 16B align)
#define YP 136                    // s pitch in y tile
#define XS_HALF (3 * 114 * XP)    // 46512 halfs = 93024 B  (total LDS)

// ---------------------------------------------------------------------------
// Prep: W (f32, OIHW 3x3) -> f16 tap-major [9][co][ci]; build Toeplitz G (f16)
// g[d] = (1-beta) * sum_{i=0..d} alpha^i * beta^(d-i)
// ---------------------------------------------------------------------------
__global__ __launch_bounds__(256) void prep_kernel(
    const float* __restrict__ Wsrc,
    _Float16* __restrict__ Wh,
    _Float16* __restrict__ Gh)
{
    const int idx = blockIdx.x * 256 + threadIdx.x;
    if (idx < 9 * CCH * CCH) {
        const int tap = idx >> 14;
        const int rem = idx & 16383;
        const int co = rem >> 7, ci = rem & 127;
        Wh[idx] = (_Float16)Wsrc[(co * CCH + ci) * 9 + tap];
    } else {
        const int k = idx - 9 * CCH * CCH;
        if (k < CCH * CCH) {
            const int t = k >> 7, s = k & 127;
            float g = 0.f;
            if (t >= s) {
                const int d = t - s;
                float sum = 0.f;
                for (int i = 0; i <= d; ++i)
                    sum += powf(ALPHA_C, (float)i) * powf(BETA_C, (float)(d - i));
                g = sum * (1.0f - BETA_C);
            }
            Gh[k] = (_Float16)g;
        }
    }
}

// ---------------------------------------------------------------------------
// Fused 3x3 conv (implicit GEMM over 9 taps) + causal temporal GEMM (mem = G@y)
// One workgroup (8 waves) per (batch, h): 128 channels x 112 pixels.
// A-fragments (W taps / G rows) come straight from L2 into registers;
// only the activation tile and the intermediate y tile live in LDS.
// ---------------------------------------------------------------------------
__global__ __launch_bounds__(256) void conv_lif_kernel(
    const float* __restrict__ x,
    const float* __restrict__ bias,
    const _Float16* __restrict__ Wh,
    const _Float16* __restrict__ Gh,
    float* __restrict__ out)
{
    __shared__ _Float16 smem[XS_HALF];
    _Float16* xs = smem;              // phase 1: [kh(3)][w(114)][ci] pitch XP
    _Float16* ys = smem;              // phase 2 alias: [w(112)][s] pitch YP

    const int tid  = threadIdx.x;
    const int wave = tid >> 5;
    const int lane = tid & 31;
    const int lo   = lane & 15;
    const int hi   = lane >> 4;
    const int blk  = blockIdx.x;
    const int b    = blk / HSP;
    const int h    = blk % HSP;
    const int m0   = wave << 4;       // this wave's 16 output rows

    // ---- stage x halo (3 rows, zero-padded width 114) into LDS as f16,
    //      4 channels per thread -> packed ds_store_b64 ----
    for (int q = wave; q < 96; q += 8) {          // 96 = 3 rows * 32 ci-groups
        const int r   = q >> 5;                   // kh row 0..2
        const int cig = q & 31;                   // group of 4 channels
        const int hr  = h - 1 + r;
        const bool hv = (hr >= 0) && (hr < HSP);
        const float* xg = x + (((size_t)(b * CCH + cig * 4)) * HSP + (hv ? hr : 0)) * WSP;
        for (int wl = lane; wl < 114; wl += 32) {
            const int wg = wl - 1;
            const bool wv = hv && (wg >= 0) && (wg < WSP);
            v4h pk;
            pk[0] = (_Float16)(wv ? xg[0 * HSP * WSP + wg] : 0.f);
            pk[1] = (_Float16)(wv ? xg[1 * HSP * WSP + wg] : 0.f);
            pk[2] = (_Float16)(wv ? xg[2 * HSP * WSP + wg] : 0.f);
            pk[3] = (_Float16)(wv ? xg[3 * HSP * WSP + wg] : 0.f);
            *(v4h*)(xs + (r * 114 + wl) * XP + cig * 4) = pk;
        }
    }

    // B-fragment loader: column = pixel (n), K = ci/s contiguous in LDS
    auto loadB = [&](const _Float16* base, int pitch, int col, int k0) -> v16h {
        v16h bf;
        const _Float16* bp = base + col * pitch + k0 + hi * 8;
        ((v8h*)&bf)[0] = *(const v8h*)bp;
        ((v8h*)&bf)[1] = *(const v8h*)(bp + 16);
        return bf;
    };
    // A-fragment loader from a global row-major [128][128] f16 matrix (L2-hot)
    auto loadA = [&](const _Float16* mat, int k0) -> v16h {
        v16h af;
        const _Float16* ap = mat + (m0 + lo) * CCH + k0 + hi * 8;
        ((v8h*)&af)[0] = *(const v8h*)ap;
        ((v8h*)&af)[1] = *(const v8h*)(ap + 16);
        return af;
    };

    v8f acc[7] = {};
    __syncthreads();

    // ---- phase 1: conv as 9 taps x (128x128 @ 128x112) GEMM ----
    #pragma unroll 1
    for (int tap = 0; tap < 9; ++tap) {
        const int kh = tap / 3;
        const int kw = tap - kh * 3;
        const _Float16* wsrc = Wh + tap * CCH * CCH;
        v16h afr[4];
        #pragma unroll
        for (int kb = 0; kb < 4; ++kb) afr[kb] = loadA(wsrc, kb * 32);   // LOADcnt
        const _Float16* xrow = xs + (kh * 114 + kw) * XP;
        #pragma unroll
        for (int kb = 0; kb < 4; ++kb) {
            const int ci0 = kb * 32;
            v16h bcur = loadB(xrow, XP, lo, ci0);                        // DScnt
            #pragma unroll
            for (int j = 0; j < 7; ++j) {
                v16h bnxt;
                if (j < 6) bnxt = loadB(xrow, XP, (j + 1) * 16 + lo, ci0);
                acc[j] = __builtin_amdgcn_wmma_f32_16x16x32_f16(
                    false, afr[kb], false, bcur, (short)0, acc[j], false, false);
                if (j < 6) bcur = bnxt;
            }
        }
    }

    // per-lane bias values for the 8 accumulator rows
    float bv[8];
    #pragma unroll
    for (int r = 0; r < 8; ++r) bv[r] = bias[m0 + hi * 8 + r];

    __syncthreads();   // everyone done reading xs before aliasing it with ys

    // ---- phase transition: y (+bias) -> LDS [w][s] in f16 ----
    #pragma unroll
    for (int j = 0; j < 7; ++j) {
        v8h pk;
        #pragma unroll
        for (int r = 0; r < 8; ++r) pk[r] = (_Float16)(acc[j][r] + bv[r]);
        *(v8h*)(ys + (j * 16 + lo) * YP + m0 + hi * 8) = pk;
    }
    __syncthreads();

    // ---- phase 2: mem = G @ y ; G lower-triangular -> skip zero K-blocks ----
    v8f acc2[7] = {};
    const int kbmax = (wave >> 1) + 1;       // blocks with s0 <= m0+15
    #pragma unroll 1
    for (int kb = 0; kb < kbmax; ++kb) {
        const int s0 = kb * 32;
        v16h afr = loadA(Gh, s0);                                        // LOADcnt
        v16h bcur = loadB(ys, YP, lo, s0);                               // DScnt
        #pragma unroll
        for (int j = 0; j < 7; ++j) {
            v16h bnxt;
            if (j < 6) bnxt = loadB(ys, YP, (j + 1) * 16 + lo, s0);
            acc2[j] = __builtin_amdgcn_wmma_f32_16x16x32_f16(
                false, afr, false, bcur, (short)0, acc2[j], false, false);
            if (j < 6) bcur = bnxt;
        }
    }

    // ---- store fp32 output, coalesced across the 16 low lanes ----
    #pragma unroll
    for (int j = 0; j < 7; ++j) {
        #pragma unroll
        for (int r = 0; r < 8; ++r) {
            const int m = m0 + hi * 8 + r;
            out[(((size_t)(b * CCH + m)) * HSP + h) * WSP + j * 16 + lo] = acc2[j][r];
        }
    }
}

extern "C" void kernel_launch(void* const* d_in, const int* in_sizes, int n_in,
                              void* d_out, int out_size, void* d_ws, size_t ws_size,
                              hipStream_t stream) {
    const float* x    = (const float*)d_in[0];
    const float* W    = (const float*)d_in[1];
    const float* bias = (const float*)d_in[2];
    float* out = (float*)d_out;

    _Float16* Wh = (_Float16*)d_ws;              // 9*128*128 halfs
    _Float16* Gh = Wh + 9 * CCH * CCH;           // 128*128 halfs

    const int prep_elems = 9 * CCH * CCH + CCH * CCH;   // 163840
    prep_kernel<<<(prep_elems + 255) / 256, 256, 0, stream>>>(W, Wh, Gh);
    conv_lif_kernel<<<BSZ * HSP, 256, 0, stream>>>(x, bias, Wh, Gh, out);
}
